// Model_85684597555802
// MI455X (gfx1250) — matrix operand order repaired and checked by reading, not compiled, not run on hardware
//
#include <hip/hip_runtime.h>
#include <hip/hip_bf16.h>

// ---------------------------------------------------------------------------
// GPT-style transformer forward (B=2,T=1024,L=12,H=12,D=768,DFF=3072,V=100256)
// Strategy: convert/transpose all weights to bf16 [N,K] once per launch, run
// every GEMM through v_wmma_f32_16x16x32_bf16 (wave32), fuse epilogues.
// Attention (2% of FLOPs) uses a streaming-softmax VALU kernel.
// ---------------------------------------------------------------------------

typedef __attribute__((ext_vector_type(16))) __bf16 v16bf;
typedef __attribute__((ext_vector_type(8)))  float  v8f;

union FragBF { uint4 u4[2]; v16bf v; };

__device__ __forceinline__ unsigned short f2bf(float f) {
    unsigned int u = __float_as_uint(f);
    unsigned int r = u + 0x7FFFu + ((u >> 16) & 1u);   // RNE
    return (unsigned short)(r >> 16);
}
__device__ __forceinline__ float bf2f(unsigned short h) {
    return __uint_as_float(((unsigned int)h) << 16);
}

// ---------------------------------------------------------------------------
// Embedding + sinusoidal positional encoding -> fp32 x[row, D]
// ---------------------------------------------------------------------------
__global__ __launch_bounds__(256) void embed_kernel(
    const int* __restrict__ idx, const float* __restrict__ emb,
    float* __restrict__ x, int Tn, int D)
{
    int row = blockIdx.x;
    int t = row % Tn;
    int token = idx[row];
    for (int d = threadIdx.x; d < D; d += 256) {
        int i2 = (d >> 1) * 2;
        float freq = __expf((float)i2 * (-9.210340371976184f / (float)D)); // -ln(1e4)/D
        float ang = (float)t * freq;
        float pe = (d & 1) ? __cosf(ang) : __sinf(ang);
        x[(size_t)row * D + d] = emb[(size_t)token * D + d] + pe;
    }
}

// ---------------------------------------------------------------------------
// LayerNorm (fp32 in) -> bf16 out. One block per row.
// ---------------------------------------------------------------------------
__global__ __launch_bounds__(256) void layernorm_kernel(
    const float* __restrict__ x, const float* __restrict__ gamma,
    const float* __restrict__ beta, unsigned short* __restrict__ out, int D)
{
    __shared__ float red[256];
    int row = blockIdx.x, tid = threadIdx.x;
    const float* xr = x + (size_t)row * D;

    float s = 0.f;
    for (int d = tid; d < D; d += 256) s += xr[d];
    red[tid] = s; __syncthreads();
    for (int st = 128; st > 0; st >>= 1) {
        if (tid < st) red[tid] += red[tid + st];
        __syncthreads();
    }
    float mu = red[0] / (float)D;
    __syncthreads();

    float s2 = 0.f;
    for (int d = tid; d < D; d += 256) { float t = xr[d] - mu; s2 += t * t; }
    red[tid] = s2; __syncthreads();
    for (int st = 128; st > 0; st >>= 1) {
        if (tid < st) red[tid] += red[tid + st];
        __syncthreads();
    }
    float rstd = rsqrtf(red[0] / (float)D + 1e-5f);
    __syncthreads();

    for (int d = tid; d < D; d += 256)
        out[(size_t)row * D + d] = f2bf((xr[d] - mu) * rstd * gamma[d] + beta[d]);
}

// ---------------------------------------------------------------------------
// fp32 [Krows, Ncols] row-major  ->  bf16 [Ncols, Krows] row-major (W^T)
// ---------------------------------------------------------------------------
__global__ __launch_bounds__(256) void transpose_f32_bf16_kernel(
    const float* __restrict__ in, unsigned short* __restrict__ out,
    int Krows, int Ncols)
{
    __shared__ float tile[32][33];
    int tx = threadIdx.x & 31;
    int ty = threadIdx.x >> 5;       // 0..7
    int k0 = blockIdx.y * 32, n0 = blockIdx.x * 32;
#pragma unroll
    for (int i = 0; i < 32; i += 8) {
        int k = k0 + ty + i, n = n0 + tx;
        float v = 0.f;
        if (k < Krows && n < Ncols) v = in[(size_t)k * Ncols + n];
        tile[ty + i][tx] = v;
    }
    __syncthreads();
#pragma unroll
    for (int i = 0; i < 32; i += 8) {
        int n = n0 + ty + i, k = k0 + tx;
        if (n < Ncols && k < Krows)
            out[(size_t)n * Krows + k] = f2bf(tile[tx][ty + i]);
    }
}

// ---------------------------------------------------------------------------
// WMMA GEMM: C[M,N] = A[M,K](bf16,row) * Bt[N,K](bf16,row)^T  + epilogue
// Block = 256 threads (8 waves), block tile 128x128, wave tile 64x32.
// Double-buffered K-loop (requires K % 64 == 0), B rows clamped for loads
// (OOB columns are computed on garbage but never stored).
// EPI: 0 = bf16 out (no bias)           [Q/K/V]
//      1 = f32 out, +bias +residual     [attn proj, MLP2] (in-place safe)
//      2 = bf16 out, +bias, ReLU        [MLP1]
//      3 = f32 out, +bias               [logits]
// ---------------------------------------------------------------------------
template <int EPI>
__global__ __launch_bounds__(256) void gemm_bf16_kernel(
    const unsigned short* __restrict__ A, const unsigned short* __restrict__ Bt,
    const float* __restrict__ bias, const float* __restrict__ resid,
    float* __restrict__ outF, unsigned short* __restrict__ outH,
    int M, int N, int K)
{
    const int lane = threadIdx.x & 31;
    const int wid  = threadIdx.x >> 5;
    const int lo = lane & 15, hi = lane >> 4;
    const int mblk = blockIdx.y * 128;
    const int nblk = blockIdx.x * 128;
    const int wm = (wid >> 2) * 64;   // 0 or 64
    const int wn = (wid & 3) * 32;    // 0,32,64,96

    // Per-lane fragment offsets (elements). A: 16x32 bf16 layout; Bt rows give
    // the B 32x16 layout (lane = output column, same K order as A).
    const int aoff0 = 8 * hi, aoff1 = 16 + 8 * hi;
    const int boff0 = 8 * hi, boff1 = 16 + 8 * hi;

    // Per-lane base pointers. B rows clamped to N-1: loads stay in-bounds and
    // garbage columns are discarded by the bounds-checked store.
    const unsigned short* ap[4];
    const unsigned short* bp[2];
#pragma unroll
    for (int i = 0; i < 4; ++i)
        ap[i] = A + (size_t)(mblk + wm + 16 * i + lo) * K;
#pragma unroll
    for (int j = 0; j < 2; ++j) {
        int br = nblk + wn + 16 * j + lo;
        br = br < N ? br : (N - 1);
        bp[j] = Bt + (size_t)br * K;
    }

    v8f acc[4][2];
#pragma unroll
    for (int i = 0; i < 4; ++i)
#pragma unroll
        for (int j = 0; j < 2; ++j)
#pragma unroll
            for (int r = 0; r < 8; ++r) acc[i][j][r] = 0.f;

    FragBF afA[4], bfA[2], afB[4], bfB[2];

#define LOAD_FRAGS(AF, BF, KOFF)                                              \
    do {                                                                      \
        _Pragma("unroll")                                                     \
        for (int i = 0; i < 4; ++i) {                                         \
            const unsigned short* p = ap[i] + (KOFF);                         \
            (AF)[i].u4[0] = *(const uint4*)(p + aoff0);                       \
            (AF)[i].u4[1] = *(const uint4*)(p + aoff1);                       \
        }                                                                     \
        _Pragma("unroll")                                                     \
        for (int j = 0; j < 2; ++j) {                                         \
            const unsigned short* p = bp[j] + (KOFF);                         \
            (BF)[j].u4[0] = *(const uint4*)(p + boff0);                       \
            (BF)[j].u4[1] = *(const uint4*)(p + boff1);                       \
        }                                                                     \
    } while (0)

#define MMA_FRAGS(AF, BF)                                                     \
    do {                                                                      \
        _Pragma("unroll")                                                     \
        for (int i = 0; i < 4; ++i)                                           \
            _Pragma("unroll")                                                 \
            for (int j = 0; j < 2; ++j)                                       \
                acc[i][j] = __builtin_amdgcn_wmma_f32_16x16x32_bf16(          \
                    false, (AF)[i].v, false, (BF)[j].v, (short)0,             \
                    acc[i][j], false, false);                                 \
    } while (0)

    // Software-pipelined K loop, 2 chunks (64 elements) per iteration.
    const int nk = K >> 5;           // number of 32-wide chunks; even for all K here
    LOAD_FRAGS(afA, bfA, 0);
    for (int t = 0; t < nk - 2; t += 2) {
        LOAD_FRAGS(afB, bfB, (t + 1) * 32);
        MMA_FRAGS(afA, bfA);
        LOAD_FRAGS(afA, bfA, (t + 2) * 32);
        MMA_FRAGS(afB, bfB);
    }
    LOAD_FRAGS(afB, bfB, (nk - 1) * 32);
    MMA_FRAGS(afA, bfA);
    MMA_FRAGS(afB, bfB);

#undef LOAD_FRAGS
#undef MMA_FRAGS

    // Epilogue. C/D layout: lane gives column n, VGPR r gives row m = r + 8*hi.
#pragma unroll
    for (int j = 0; j < 2; ++j) {
        int n = nblk + wn + 16 * j + lo;
        if (n >= N) continue;
        float bv = (EPI != 0) ? bias[n] : 0.f;
#pragma unroll
        for (int i = 0; i < 4; ++i) {
#pragma unroll
            for (int r = 0; r < 8; ++r) {
                int m = mblk + wm + 16 * i + 8 * hi + r;
                size_t idxo = (size_t)m * N + n;
                float v = acc[i][j][r] + bv;
                if (EPI == 1) v += resid[idxo];
                if (EPI == 2) v = fmaxf(v, 0.f);
                if (EPI == 0 || EPI == 2) outH[idxo] = f2bf(v);
                else                      outF[idxo] = v;
            }
        }
    }
}

// ---------------------------------------------------------------------------
// Causal attention, streaming softmax. One wave per (b,h,query).
// q/k/v/o are bf16 [B*T, H*64] (head-concat layout from the QKV GEMMs).
// Lane covers dims {lane, lane+32} of DH=64.
// ---------------------------------------------------------------------------
__global__ __launch_bounds__(256) void attention_kernel(
    const unsigned short* __restrict__ q, const unsigned short* __restrict__ k,
    const unsigned short* __restrict__ v, unsigned short* __restrict__ o,
    int Bn, int Hn, int Tn)
{
    int w = blockIdx.x * 8 + (threadIdx.x >> 5);
    int lane = threadIdx.x & 31;
    int total = Bn * Hn * Tn;
    if (w >= total) return;
    int tq = w % Tn;
    int bh = w / Tn;
    int h = bh % Hn;
    int b = bh / Hn;
    int ld = Hn * 64;
    size_t qbase = (size_t)(b * Tn + tq) * ld + h * 64;

    float q0 = bf2f(q[qbase + lane]);
    float q1 = bf2f(q[qbase + 32 + lane]);
    float mrun = -__builtin_inff(), lrun = 0.f, a0 = 0.f, a1 = 0.f;
    const float scale = 0.125f;  // 1/sqrt(64)

    for (int j = 0; j <= tq; ++j) {
        size_t kbse = (size_t)(b * Tn + j) * ld + h * 64;
        float s = q0 * bf2f(k[kbse + lane]) + q1 * bf2f(k[kbse + 32 + lane]);
#pragma unroll
        for (int off = 16; off > 0; off >>= 1) s += __shfl_xor(s, off, 32);
        s *= scale;
        float mnew = fmaxf(mrun, s);
        float c = __expf(mrun - mnew);
        float p = __expf(s - mnew);
        lrun = lrun * c + p;
        a0 = a0 * c + p * bf2f(v[kbse + lane]);
        a1 = a1 * c + p * bf2f(v[kbse + 32 + lane]);
        mrun = mnew;
    }
    float inv = 1.f / lrun;
    o[qbase + lane]      = f2bf(a0 * inv);
    o[qbase + 32 + lane] = f2bf(a1 * inv);
}

// ---------------------------------------------------------------------------
// Per-row loss: lse(logits_row) - logits_row[target]. One-pass online logsumexp.
// ---------------------------------------------------------------------------
__global__ __launch_bounds__(256) void loss_row_kernel(
    const float* __restrict__ logits, const int* __restrict__ targets,
    float* __restrict__ row_loss, int V)
{
    __shared__ float sm[256], ss[256];
    int row = blockIdx.x, tid = threadIdx.x;
    const float* lr = logits + (size_t)row * V;
    float m = -__builtin_inff(), s = 0.f;
    for (int j = tid; j < V; j += 256) {
        float xv = lr[j];
        if (xv > m) { s = s * __expf(m - xv) + 1.f; m = xv; }
        else        { s += __expf(xv - m); }
    }
    sm[tid] = m; ss[tid] = s; __syncthreads();
    for (int st = 128; st > 0; st >>= 1) {
        if (tid < st) {
            float m2 = sm[tid + st], s2 = ss[tid + st];
            float mN = fmaxf(sm[tid], m2);
            ss[tid] = ss[tid] * __expf(sm[tid] - mN) + s2 * __expf(m2 - mN);
            sm[tid] = mN;
        }
        __syncthreads();
    }
    if (tid == 0) {
        float lse = sm[0] + logf(ss[0]);
        row_loss[row] = lse - lr[targets[row]];
    }
}

__global__ __launch_bounds__(256) void loss_reduce_kernel(
    const float* __restrict__ row_loss, float* __restrict__ out, int n)
{
    __shared__ float red[256];
    float s = 0.f;
    for (int i = threadIdx.x; i < n; i += 256) s += row_loss[i];
    red[threadIdx.x] = s; __syncthreads();
    for (int st = 128; st > 0; st >>= 1) {
        if (threadIdx.x < st) red[threadIdx.x] += red[threadIdx.x + st];
        __syncthreads();
    }
    if (threadIdx.x == 0) *out = red[0] / (float)n;
}

// ---------------------------------------------------------------------------
// Host-side orchestration
// ---------------------------------------------------------------------------
extern "C" void kernel_launch(void* const* d_in, const int* in_sizes, int n_in,
                              void* d_out, int out_size, void* d_ws, size_t ws_size,
                              hipStream_t stream)
{
    const int Lc = 12, Hc = 12, Dc = 768, DFF = 3072, Vc = 100256;
    const int Bc = 2, Tc = 1024, Mr = Bc * Tc;   // 2048 rows

    const int*   idx     = (const int*)  d_in[0];
    const int*   targets = (const int*)  d_in[1];
    const float* emb     = (const float*)d_in[2];
    const float* wq      = (const float*)d_in[3];
    const float* wk      = (const float*)d_in[4];
    const float* wv      = (const float*)d_in[5];
    const float* wo      = (const float*)d_in[6];
    const float* bo      = (const float*)d_in[7];
    const float* ln1_s   = (const float*)d_in[8];
    const float* ln1_b   = (const float*)d_in[9];
    const float* ln2_s   = (const float*)d_in[10];
    const float* ln2_b   = (const float*)d_in[11];
    const float* w1      = (const float*)d_in[12];
    const float* b1      = (const float*)d_in[13];
    const float* w2      = (const float*)d_in[14];
    const float* b2      = (const float*)d_in[15];
    const float* lnf_s   = (const float*)d_in[16];
    const float* lnf_b   = (const float*)d_in[17];
    const float* w_out   = (const float*)d_in[18];
    const float* b_out   = (const float*)d_in[19];

    float* logits   = (float*)d_out;
    float* loss_out = logits + (size_t)Mr * Vc;

    // Workspace carve-out (~205 MB), 256B aligned.
    char* wsb = (char*)d_ws;
    size_t off = 0;
    auto alloc = [&](size_t bytes) -> void* {
        void* p = wsb + off;
        off = (off + bytes + 255) & ~(size_t)255;
        return p;
    };
    unsigned short* wqT   = (unsigned short*)alloc((size_t)Dc * Dc * 2);
    unsigned short* wkT   = (unsigned short*)alloc((size_t)Dc * Dc * 2);
    unsigned short* wvT   = (unsigned short*)alloc((size_t)Dc * Dc * 2);
    unsigned short* woT   = (unsigned short*)alloc((size_t)Dc * Dc * 2);
    unsigned short* w1T   = (unsigned short*)alloc((size_t)Dc * DFF * 2);
    unsigned short* w2T   = (unsigned short*)alloc((size_t)Dc * DFF * 2);
    unsigned short* woutT = (unsigned short*)alloc((size_t)Vc * Dc * 2);
    float*          x     = (float*)        alloc((size_t)Mr * Dc * 4);
    unsigned short* hb    = (unsigned short*)alloc((size_t)Mr * Dc * 2);
    unsigned short* qb    = (unsigned short*)alloc((size_t)Mr * Dc * 2);
    unsigned short* kb    = (unsigned short*)alloc((size_t)Mr * Dc * 2);
    unsigned short* vb    = (unsigned short*)alloc((size_t)Mr * Dc * 2);
    unsigned short* ob    = (unsigned short*)alloc((size_t)Mr * Dc * 2);
    unsigned short* midb  = (unsigned short*)alloc((size_t)Mr * DFF * 2);
    float*          rloss = (float*)        alloc((size_t)Mr * 4);

    dim3 blk(256);
    dim3 gT_DD(Dc / 32, Dc / 32);           // 768x768 transpose
    dim3 gT_W1(DFF / 32, Dc / 32);          // [D,DFF] -> [DFF,D]
    dim3 gT_W2(Dc / 32, DFF / 32);          // [DFF,D] -> [D,DFF]
    dim3 gG_D(Dc / 128, Mr / 128);          // (6,16)
    dim3 gG_F(DFF / 128, Mr / 128);         // (24,16)
    dim3 gG_V((Vc + 127) / 128, Mr / 128);  // (784,16)

    embed_kernel<<<Mr, blk, 0, stream>>>(idx, emb, x, Tc, Dc);

    for (int l = 0; l < Lc; ++l) {
        transpose_f32_bf16_kernel<<<gT_DD, blk, 0, stream>>>(wq + (size_t)l * Dc * Dc, wqT, Dc, Dc);
        transpose_f32_bf16_kernel<<<gT_DD, blk, 0, stream>>>(wk + (size_t)l * Dc * Dc, wkT, Dc, Dc);
        transpose_f32_bf16_kernel<<<gT_DD, blk, 0, stream>>>(wv + (size_t)l * Dc * Dc, wvT, Dc, Dc);
        transpose_f32_bf16_kernel<<<gT_DD, blk, 0, stream>>>(wo + (size_t)l * Dc * Dc, woT, Dc, Dc);
        transpose_f32_bf16_kernel<<<gT_W1, blk, 0, stream>>>(w1 + (size_t)l * Dc * DFF, w1T, Dc, DFF);
        transpose_f32_bf16_kernel<<<gT_W2, blk, 0, stream>>>(w2 + (size_t)l * DFF * Dc, w2T, DFF, Dc);

        layernorm_kernel<<<Mr, blk, 0, stream>>>(x, ln1_s + l * Dc, ln1_b + l * Dc, hb, Dc);

        gemm_bf16_kernel<0><<<gG_D, blk, 0, stream>>>(hb, wqT, nullptr, nullptr, nullptr, qb, Mr, Dc, Dc);
        gemm_bf16_kernel<0><<<gG_D, blk, 0, stream>>>(hb, wkT, nullptr, nullptr, nullptr, kb, Mr, Dc, Dc);
        gemm_bf16_kernel<0><<<gG_D, blk, 0, stream>>>(hb, wvT, nullptr, nullptr, nullptr, vb, Mr, Dc, Dc);

        attention_kernel<<<(Bc * Hc * Tc) / 8, blk, 0, stream>>>(qb, kb, vb, ob, Bc, Hc, Tc);

        // x = x + o @ wo + bo   (in-place residual: each element read+written by same thread)
        gemm_bf16_kernel<1><<<gG_D, blk, 0, stream>>>(ob, woT, bo + l * Dc, x, x, nullptr, Mr, Dc, Dc);

        layernorm_kernel<<<Mr, blk, 0, stream>>>(x, ln2_s + l * Dc, ln2_b + l * Dc, hb, Dc);

        gemm_bf16_kernel<2><<<gG_F, blk, 0, stream>>>(hb, w1T, b1 + l * DFF, nullptr, nullptr, midb, Mr, DFF, Dc);
        gemm_bf16_kernel<1><<<gG_D, blk, 0, stream>>>(midb, w2T, b2 + l * Dc, x, x, nullptr, Mr, Dc, DFF);
    }

    layernorm_kernel<<<Mr, blk, 0, stream>>>(x, lnf_s, lnf_b, hb, Dc);

    transpose_f32_bf16_kernel<<<dim3(Vc / 32, Dc / 32), blk, 0, stream>>>(w_out, woutT, Dc, Vc);
    gemm_bf16_kernel<3><<<gG_V, blk, 0, stream>>>(hb, woutT, b_out, nullptr, logits, nullptr, Mr, Vc, Dc);

    loss_row_kernel<<<Mr, blk, 0, stream>>>(logits, targets, rloss, Vc);
    loss_reduce_kernel<<<1, blk, 0, stream>>>(rloss, loss_out, Mr);
}
